// LongRangeInteraction_90829968376327
// MI455X (gfx1250) — matrix operand order repaired and checked by reading, not compiled, not run on hardware
//
#include <hip/hip_runtime.h>
#include <hip/hip_bf16.h>

typedef __attribute__((ext_vector_type(16))) _Float16 v16h;
typedef __attribute__((ext_vector_type(8)))  float    v8f;

#define N_ATOMS 8192
#define NB      32
#define NK      128
#define ND      128
#define MROWS   (NB * NK)   // 4096 rows of the k-space MLP

// ---------------------------------------------------------------- helpers

__device__ __forceinline__ float gelu_f(float x) {
  // tanh-approx GELU (jax.nn.gelu default)
  float x3 = x * x * x;
  return 0.5f * x * (1.0f + tanhf(0.7978845608028654f * (x + 0.044715f * x3)));
}

__device__ __forceinline__ v8f wmma_f16(v16h a, v16h b, v8f c) {
  // D = A(16x32 f16) * B(32x16 f16) + C(16x16 f32)
  return __builtin_amdgcn_wmma_f32_16x16x32_f16(
      /*neg_a=*/false, a, /*neg_b=*/false, b,
      /*c_mod=*/(short)0, c, /*reuse_a=*/false, /*reuse_b=*/false);
}

// A fragment (16x32, f16) from a ROW-MAJOR matrix: row = r0 + lane%16,
// k-range [k0, k0+32). Lane<16 holds k-run {0..15}, lane>=16 {16..31}.
// Rows >= rowEnd are zero-padded (keeps EXEC all-ones for the WMMA itself).
__device__ __forceinline__ v16h load_a_rowmajor(const _Float16* base, int r0,
                                                int rowEnd, int ld, int k0,
                                                int lane) {
  int m  = r0 + (lane & 15);
  int kb = (lane < 16) ? 0 : 16;
  v16h a;
  if (m < rowEnd) {
    const _Float16* p = base + (size_t)m * ld + k0 + kb;
#pragma unroll
    for (int e = 0; e < 8; ++e) a[e] = p[e];
#pragma unroll
    for (int e = 0; e < 8; ++e) a[8 + e] = p[8 + e];
  } else {
#pragma unroll
    for (int e = 0; e < 16; ++e) a[e] = (_Float16)0.0f;
  }
  return a;
}

// A fragment (16x32, f16) where the 32-wide K-dim indexes ATOMS in a
// transposed matrix T[k][atom] (ld = N_ATOMS). Atom columns beyond segEnd -> 0.
__device__ __forceinline__ v16h load_a_transposed(const _Float16* base, int krow0,
                                                  int a0, int segEnd, int lane) {
  int k  = krow0 + (lane & 15);
  int ab = (lane < 16) ? 0 : 16;
  const _Float16* p = base + (size_t)k * N_ATOMS + a0;
  v16h a;
#pragma unroll
  for (int e = 0; e < 8; ++e) {
    int c0 = ab + e;
    a[e] = (a0 + c0 < segEnd) ? p[c0] : (_Float16)0.0f;
  }
#pragma unroll
  for (int e = 0; e < 8; ++e) {
    int c1 = ab + 8 + e;
    a[8 + e] = (a0 + c1 < segEnd) ? p[c1] : (_Float16)0.0f;
  }
  return a;
}

// B fragment (32x16, f16) from a COLUMN-MAJOR matrix BT[n][k] (ld halfs per
// column). Lane holds column n0+lane%16; k-run = k0 + (lane<16?0:16) + e.
__device__ __forceinline__ v16h load_b_colmajor(const _Float16* bt, int n0,
                                                int ld, int k0, int lane) {
  int n  = n0 + (lane & 15);
  int kb = (lane < 16) ? 0 : 16;
  const _Float16* p = bt + (size_t)n * ld + k0 + kb;
  v16h b;
#pragma unroll
  for (int e = 0; e < 16; ++e) b[e] = p[e];
  return b;
}

// Same but K indexes atoms with zero-padding beyond segEnd (HT[d][atom]).
__device__ __forceinline__ v16h load_b_atoms(const _Float16* ht, int d0, int a0,
                                             int segEnd, int lane) {
  int d  = d0 + (lane & 15);
  int kb = (lane < 16) ? 0 : 16;
  const _Float16* p = ht + (size_t)d * N_ATOMS + a0;
  v16h b;
#pragma unroll
  for (int e = 0; e < 16; ++e) {
    int c = kb + e;
    b[e] = (a0 + c < segEnd) ? p[c] : (_Float16)0.0f;
  }
  return b;
}

// B fragment (32x16, f16) from an LDS-resident column-major tile
// lds[16][NK] (row = output column d-local, contiguous in k).
__device__ __forceinline__ v16h load_b_lds(const _Float16* lds, int k0, int lane) {
  int n  = lane & 15;
  int kb = (lane < 16) ? 0 : 16;
  const _Float16* p = lds + (size_t)n * NK + k0 + kb;
  v16h b;
#pragma unroll
  for (int e = 0; e < 16; ++e) b[e] = p[e];
  return b;
}

// ---------------------------------------------------------------- kernels

// Segment offsets: seg[b] = lower_bound(batch, b), seg[NB] = N_ATOMS.
__global__ void lri_seg_kernel(const int* __restrict__ batch, int* __restrict__ seg) {
  int b = blockIdx.x * blockDim.x + threadIdx.x;
  if (b > NB) return;
  int lo = 0, hi = N_ATOMS;
  while (lo < hi) {
    int mid = (lo + hi) >> 1;
    if (batch[mid] < b) lo = mid + 1; else hi = mid;
  }
  seg[b] = lo;
}

// theta, cos, sin -> f16 in both row-major [N][K] and transposed [K][N].
__global__ void lri_phase_kernel(const float* __restrict__ kvec,
                                 const float* __restrict__ pos,
                                 const int* __restrict__ batch,
                                 _Float16* __restrict__ crow,
                                 _Float16* __restrict__ srow,
                                 _Float16* __restrict__ ct,
                                 _Float16* __restrict__ st) {
  int idx = blockIdx.x * blockDim.x + threadIdx.x;
  if (idx >= N_ATOMS * NK) return;
  int n = idx / NK, k = idx % NK;
  int b = batch[n];
  const float* kv = kvec + ((size_t)(b * NK + k)) * 3;
  const float* pr = pos + (size_t)n * 3;
  float theta = pr[0] * kv[0] + pr[1] * kv[1] + pr[2] * kv[2];
  float c = cosf(theta), s = sinf(theta);
  crow[(size_t)n * NK + k] = (_Float16)c;
  srow[(size_t)n * NK + k] = (_Float16)s;
  ct[(size_t)k * N_ATOMS + n] = (_Float16)c;
  st[(size_t)k * N_ATOMS + n] = (_Float16)s;
}

// h [N][D] f32 -> HT [D][N] f16
__global__ void lri_ht_kernel(const float* __restrict__ h, _Float16* __restrict__ ht) {
  int idx = blockIdx.x * blockDim.x + threadIdx.x;
  if (idx >= N_ATOMS * ND) return;
  int n = idx / ND, d = idx % ND;
  ht[(size_t)d * N_ATOMS + n] = (_Float16)h[(size_t)n * ND + d];
}

// W2, W3 [K][N] f32 -> column-major f16 (WT[n][k])
__global__ void lri_wconv_kernel(const float* __restrict__ w2,
                                 const float* __restrict__ w3,
                                 _Float16* __restrict__ w2t,
                                 _Float16* __restrict__ w3t) {
  int idx = blockIdx.x * blockDim.x + threadIdx.x;
  if (idx >= ND * ND) return;
  int k = idx / ND, n = idx % ND;
  w2t[(size_t)n * ND + k] = (_Float16)w2[(size_t)k * ND + n];
  w3t[(size_t)n * ND + k] = (_Float16)w3[(size_t)k * ND + n];
}

// MLP layer 1 (3 -> 128) + GELU, scalar (K=3 is too small for WMMA).
__global__ void lri_mlp1_kernel(const float* __restrict__ kvec,
                                const float* __restrict__ W1,
                                const float* __restrict__ b1,
                                _Float16* __restrict__ a1) {
  int idx = blockIdx.x * blockDim.x + threadIdx.x;
  if (idx >= MROWS * ND) return;
  int m = idx / ND, j = idx % ND;
  const float* kv = kvec + (size_t)m * 3;
  float x = kv[0] * W1[j] + kv[1] * W1[ND + j] + kv[2] * W1[2 * ND + j] + b1[j];
  a1[idx] = (_Float16)gelu_f(x);
}

// Generic WMMA GEMM: A [MROWS][128] f16 row-major  x  BT [128][128] f16
// column-major + bias; mode 0: GELU -> f16 out; mode 1: plain -> f32 out.
__global__ void lri_mlp_gemm_kernel(const _Float16* __restrict__ A,
                                    const _Float16* __restrict__ BT,
                                    const float* __restrict__ bias,
                                    _Float16* __restrict__ out16,
                                    float* __restrict__ out32, int mode) {
  int lane = threadIdx.x;
  int m0 = blockIdx.x * 16;
  int n0 = blockIdx.y * 16;
  v8f acc = {};
#pragma unroll
  for (int kc = 0; kc < ND / 32; ++kc) {
    v16h a = load_a_rowmajor(A, m0, MROWS, ND, kc * 32, lane);
    v16h b = load_b_colmajor(BT, n0, ND, kc * 32, lane);
    acc = wmma_f16(a, b, acc);
  }
  int col = n0 + (lane & 15);
  float bv = bias[col];
#pragma unroll
  for (int r = 0; r < 8; ++r) {
    int m = m0 + ((lane < 16) ? r : 8 + r);
    float v = acc[r] + bv;
    if (mode == 0) out16[(size_t)m * ND + col] = (_Float16)gelu_f(v);
    else           out32[(size_t)m * ND + col] = v;
  }
}

// Segmented scatter GEMM: S_re[b,k,d] = sum_{n in b} cos * h,
//                         S_im[b,k,d] = sum_{n in b} -sin * h.
// Grid: (K/32, NB, D/16); one wave per 32x16 (k,d) tile of one batch
// (two 16-row k-tiles share one h^T B-fragment per 32-atom chunk).
__global__ void lri_scatter_kernel(const _Float16* __restrict__ ct,
                                   const _Float16* __restrict__ st,
                                   const _Float16* __restrict__ ht,
                                   const int* __restrict__ seg,
                                   float* __restrict__ sre,
                                   float* __restrict__ sim) {
  int lane = threadIdx.x;
  int k0 = blockIdx.x * 32;
  int b  = blockIdx.y;
  int d0 = blockIdx.z * 16;
  int s0 = seg[b], s1 = seg[b + 1];
  v8f accre0 = {}, accim0 = {}, accre1 = {}, accim1 = {};
  for (int a0 = s0; a0 < s1; a0 += 32) {
    v16h bh  = load_b_atoms(ht, d0, a0, s1, lane);       // shared by both k-tiles
    v16h ac0 = load_a_transposed(ct, k0,      a0, s1, lane);
    v16h as0 = load_a_transposed(st, k0,      a0, s1, lane);
    v16h ac1 = load_a_transposed(ct, k0 + 16, a0, s1, lane);
    v16h as1 = load_a_transposed(st, k0 + 16, a0, s1, lane);
    v16h asn0 = -as0;                    // f16 WMMA has no A-negate modifier
    v16h asn1 = -as1;
    accre0 = wmma_f16(ac0,  bh, accre0);
    accim0 = wmma_f16(asn0, bh, accim0);
    accre1 = wmma_f16(ac1,  bh, accre1);
    accim1 = wmma_f16(asn1, bh, accim1);
  }
  int d = d0 + (lane & 15);
#pragma unroll
  for (int r = 0; r < 8; ++r) {
    int kk = k0 + ((lane < 16) ? r : 8 + r);
    size_t o0 = ((size_t)b * NK + kk) * ND + d;
    size_t o1 = ((size_t)b * NK + kk + 16) * ND + d;
    sre[o0] = accre0[r];
    sim[o0] = accim0[r];
    sre[o1] = accre1[r];
    sim[o1] = accim1[r];
  }
}

// G = F * S (elementwise), stored transposed per batch: GT[b][d][k] f16.
__global__ void lri_gt_kernel(const float* __restrict__ F,
                              const float* __restrict__ sre,
                              const float* __restrict__ sim,
                              _Float16* __restrict__ gtre,
                              _Float16* __restrict__ gtim) {
  int idx = blockIdx.x * blockDim.x + threadIdx.x;
  if (idx >= NB * NK * ND) return;
  int b = idx / (NK * ND);
  int r = idx % (NK * ND);
  int k = r / ND, d = r % ND;
  float f = F[idx];
  size_t o = ((size_t)b * ND + d) * NK + k;
  gtre[o] = (_Float16)(f * sre[idx]);
  gtim[o] = (_Float16)(f * sim[idx]);
}

// Back-projection: out[n,d] = sum_k exp(+i theta) * G[b_n,k,d], complex out.
// Grid: (D/16, NB); 4-wave workgroups. The (b, d-tile) G slab (16x128 re+im,
// 8 KB f16) is loop-invariant across atom tiles, so stage it in LDS once and
// let all 4 waves stream atom tiles against it via ds_load.
__global__ void lri_gather_kernel(const _Float16* __restrict__ crow,
                                  const _Float16* __restrict__ srow,
                                  const _Float16* __restrict__ gtre,
                                  const _Float16* __restrict__ gtim,
                                  const int* __restrict__ seg,
                                  float* __restrict__ out) {
  __shared__ __align__(16) _Float16 ldsre[16 * NK];
  __shared__ __align__(16) _Float16 ldsim[16 * NK];

  int tid  = threadIdx.x;
  int lane = tid & 31;
  int wv   = tid >> 5;          // 0..3
  int d0 = blockIdx.x * 16;
  int b  = blockIdx.y;
  int s0 = seg[b], s1 = seg[b + 1];

  // Stage G tile: rows d0..d0+15 of GT[b] are contiguous (16*NK halfs each).
  {
    const uint4* srcre = (const uint4*)(gtre + ((size_t)b * ND + d0) * NK);
    const uint4* srcim = (const uint4*)(gtim + ((size_t)b * ND + d0) * NK);
    uint4* dstre = (uint4*)ldsre;
    uint4* dstim = (uint4*)ldsim;
    const int n16 = 16 * NK * (int)sizeof(_Float16) / (int)sizeof(uint4); // 256
#pragma unroll
    for (int i = tid; i < n16; i += 128) {
      dstre[i] = srcre[i];
      dstim[i] = srcim[i];
    }
  }
  __syncthreads();

  int ntiles = (s1 - s0 + 15) >> 4;
  int dcol = d0 + (lane & 15);

  for (int t = wv; t < ntiles; t += 4) {
    int m0 = s0 + t * 16;
    v8f accre = {}, accim = {};
#pragma unroll
    for (int kc = 0; kc < NK / 32; ++kc) {
      int k0 = kc * 32;
      v16h ac = load_a_rowmajor(crow, m0, s1, NK, k0, lane);
      v16h as = load_a_rowmajor(srow, m0, s1, NK, k0, lane);
      v16h asn = -as;
      v16h bre = load_b_lds(ldsre, k0, lane);
      v16h bim = load_b_lds(ldsim, k0, lane);
      accre = wmma_f16(ac,  bre, accre);   //  c * G_re
      accre = wmma_f16(asn, bim, accre);   // -s * G_im
      accim = wmma_f16(as,  bre, accim);   //  s * G_re
      accim = wmma_f16(ac,  bim, accim);   //  c * G_im
    }
#pragma unroll
    for (int r = 0; r < 8; ++r) {
      int m = m0 + ((lane < 16) ? r : 8 + r);
      if (m < s1) {
        size_t o = ((size_t)m * ND + dcol) * 2;  // complex64 interleaved
        out[o]     = accre[r];
        out[o + 1] = accim[r];
      }
    }
  }
}

// ---------------------------------------------------------------- launch

extern "C" void kernel_launch(void* const* d_in, const int* in_sizes, int n_in,
                              void* d_out, int out_size, void* d_ws, size_t ws_size,
                              hipStream_t stream) {
  const float* kvec = (const float*)d_in[0];
  const float* pos  = (const float*)d_in[1];
  const float* h    = (const float*)d_in[2];
  const float* W1   = (const float*)d_in[3];
  const float* b1   = (const float*)d_in[4];
  const float* W2   = (const float*)d_in[5];
  const float* b2   = (const float*)d_in[6];
  const float* W3   = (const float*)d_in[7];
  const float* b3   = (const float*)d_in[8];
  const int* batch  = (const int*)d_in[9];
  float* out = (float*)d_out;

  char* ws = (char*)d_ws;
  size_t off = 0;
  auto alloc = [&](size_t bytes) {
    void* p = ws + off;
    off = (off + bytes + 255) & ~(size_t)255;
    return p;
  };
  int*       seg  = (int*)alloc((NB + 1) * sizeof(int));
  _Float16*  crow = (_Float16*)alloc((size_t)N_ATOMS * NK * 2);
  _Float16*  srow = (_Float16*)alloc((size_t)N_ATOMS * NK * 2);
  _Float16*  ct   = (_Float16*)alloc((size_t)NK * N_ATOMS * 2);
  _Float16*  st   = (_Float16*)alloc((size_t)NK * N_ATOMS * 2);
  _Float16*  ht   = (_Float16*)alloc((size_t)ND * N_ATOMS * 2);
  _Float16*  w2t  = (_Float16*)alloc((size_t)ND * ND * 2);
  _Float16*  w3t  = (_Float16*)alloc((size_t)ND * ND * 2);
  _Float16*  a1   = (_Float16*)alloc((size_t)MROWS * ND * 2);
  _Float16*  a2   = (_Float16*)alloc((size_t)MROWS * ND * 2);
  float*     Fbuf = (float*)alloc((size_t)MROWS * ND * 4);
  float*     sre  = (float*)alloc((size_t)NB * NK * ND * 4);
  float*     sim  = (float*)alloc((size_t)NB * NK * ND * 4);
  _Float16*  gtre = (_Float16*)alloc((size_t)NB * ND * NK * 2);
  _Float16*  gtim = (_Float16*)alloc((size_t)NB * ND * NK * 2);

  lri_seg_kernel<<<1, 64, 0, stream>>>(batch, seg);

  lri_phase_kernel<<<(N_ATOMS * NK + 255) / 256, 256, 0, stream>>>(
      kvec, pos, batch, crow, srow, ct, st);

  lri_ht_kernel<<<(N_ATOMS * ND + 255) / 256, 256, 0, stream>>>(h, ht);

  lri_wconv_kernel<<<(ND * ND + 255) / 256, 256, 0, stream>>>(W2, W3, w2t, w3t);

  lri_mlp1_kernel<<<(MROWS * ND + 255) / 256, 256, 0, stream>>>(kvec, W1, b1, a1);

  lri_mlp_gemm_kernel<<<dim3(MROWS / 16, ND / 16), 32, 0, stream>>>(
      a1, w2t, b2, a2, (float*)nullptr, 0);
  lri_mlp_gemm_kernel<<<dim3(MROWS / 16, ND / 16), 32, 0, stream>>>(
      a2, w3t, b3, (_Float16*)nullptr, Fbuf, 1);

  lri_scatter_kernel<<<dim3(NK / 32, NB, ND / 16), 32, 0, stream>>>(
      ct, st, ht, seg, sre, sim);

  lri_gt_kernel<<<(NB * NK * ND + 255) / 256, 256, 0, stream>>>(
      Fbuf, sre, sim, gtre, gtim);

  lri_gather_kernel<<<dim3(ND / 16, NB), 128, 0, stream>>>(
      crow, srow, gtre, gtim, seg, out);
}
